// DeformableTransformerEncoderLayer_34806414967062
// MI455X (gfx1250) — matrix operand order repaired and checked by reading, not compiled, not run on hardware
//
#include <hip/hip_runtime.h>

typedef __attribute__((ext_vector_type(16))) __bf16 v16bf;
typedef __attribute__((ext_vector_type(8)))  __bf16 v8bf;
typedef __attribute__((ext_vector_type(8)))  float  v8f;
typedef __attribute__((ext_vector_type(4)))  int    v4i;

static constexpr int LQc   = 19560;
static constexpr int DMOD  = 256;
static constexpr int DFFN  = 1024;

__device__ __forceinline__ unsigned short f2bf(float f) {
  unsigned int u = __float_as_uint(f);
  u += 0x7fffu + ((u >> 16) & 1u);        // round-to-nearest-even
  return (unsigned short)(u >> 16);
}

// ---- CDNA5 async global->LDS copy (ASYNCcnt) with compile-safe fallback ----
#if __has_builtin(__builtin_amdgcn_global_load_async_to_lds_b128)
#define HAVE_ASYNC_LDS 1
#else
#define HAVE_ASYNC_LDS 0
#endif

__device__ __forceinline__ void cp16(unsigned short* l, const unsigned short* g) {
#if HAVE_ASYNC_LDS
  // generic LDS pointer low 32 bits == LDS byte offset (flat shared aperture)
  unsigned loff = (unsigned)(unsigned long long)l;
  __builtin_amdgcn_global_load_async_to_lds_b128(
      (__attribute__((address_space(1))) v4i*)(unsigned long long)g,
      (__attribute__((address_space(3))) v4i*)loff, 0, 0);
#else
  *(uint4*)l = *(const uint4*)g;
#endif
}

#if HAVE_ASYNC_LDS
#if __has_builtin(__builtin_amdgcn_s_wait_asynccnt)
#define AWAIT(n) __builtin_amdgcn_s_wait_asynccnt(n)
#else
#define AWAIT(n) asm volatile("s_wait_asynccnt " #n ::: "memory")
#endif
#else
#define AWAIT(n) ((void)0)
#endif

// ---------------- weight convert + transpose: W[K,N] f32 -> Wt[N,K] bf16 ----
__global__ __launch_bounds__(256) void wt_cvt_kernel(const float* __restrict__ W,
                                                     unsigned short* __restrict__ Wt,
                                                     int K, int N) {
  int idx = blockIdx.x * blockDim.x + threadIdx.x;
  if (idx >= K * N) return;
  int n = idx / K, k = idx - n * K;
  Wt[idx] = f2bf(W[(size_t)k * N + n]);
}

// ---------------- elementwise (a [+ b]) -> f32 and/or bf16 ------------------
__global__ __launch_bounds__(256) void add_cvt_kernel(const float* __restrict__ a,
                                                      const float* __restrict__ b,
                                                      float* __restrict__ of,
                                                      unsigned short* __restrict__ oh,
                                                      int n4) {
  int i = blockIdx.x * blockDim.x + threadIdx.x;
  if (i >= n4) return;
  size_t o = (size_t)i * 4;
  float4 va = *(const float4*)(a + o);
  if (b) {
    float4 vb = *(const float4*)(b + o);
    va.x += vb.x; va.y += vb.y; va.z += vb.z; va.w += vb.w;
  }
  if (of) *(float4*)(of + o) = va;
  if (oh) {
    ushort4 u; u.x = f2bf(va.x); u.y = f2bf(va.y); u.z = f2bf(va.z); u.w = f2bf(va.w);
    *(ushort4*)(oh + o) = u;
  }
}

// ---------------- layernorm: out = LN(a [+ resid]) * g + b ------------------
__global__ __launch_bounds__(256) void ln_kernel(const float* __restrict__ a,
                                                 const float* __restrict__ resid,
                                                 const float* __restrict__ g,
                                                 const float* __restrict__ bta,
                                                 float* __restrict__ out,
                                                 unsigned short* __restrict__ outb,
                                                 int M) {
  int row  = blockIdx.x * 8 + (threadIdx.x >> 5);
  int lane = threadIdx.x & 31;
  if (row >= M) return;
  const float* pa = a + (size_t)row * DMOD;
  const float* pr = resid ? resid + (size_t)row * DMOD : nullptr;
  float x[8];
  float s = 0.f;
#pragma unroll
  for (int i = 0; i < 8; i++) {
    float v = pa[i * 32 + lane];
    if (pr) v += pr[i * 32 + lane];
    x[i] = v; s += v;
  }
#pragma unroll
  for (int t = 16; t >= 1; t >>= 1) s += __shfl_xor(s, t, 32);
  float mean = s * (1.f / 256.f);
  float vs = 0.f;
#pragma unroll
  for (int i = 0; i < 8; i++) { float d = x[i] - mean; vs += d * d; }
#pragma unroll
  for (int t = 16; t >= 1; t >>= 1) vs += __shfl_xor(vs, t, 32);
  float rstd = rsqrtf(vs * (1.f / 256.f) + 1e-5f);
#pragma unroll
  for (int i = 0; i < 8; i++) {
    int c = i * 32 + lane;
    float y = (x[i] - mean) * rstd * g[c] + bta[c];
    out[(size_t)row * DMOD + c] = y;
    if (outb) outb[(size_t)row * DMOD + c] = f2bf(y);
  }
}

// ---------------- WMMA GEMM: C[M,N] = A_bf16[M,K] @ Wt_bf16[N,K]^T + bias ---
// block tile 128(M) x 64(N); 8 waves, each wave = 2x2 grid of 16x16 WMMA tiles
// double-buffered LDS, async global->LDS copies when available
__global__ __launch_bounds__(256) void gemm_bf16_wmma(const unsigned short* __restrict__ A,
                                                      const unsigned short* __restrict__ Wt,
                                                      const float* __restrict__ bias,
                                                      float* __restrict__ Cf,
                                                      unsigned short* __restrict__ Cb,
                                                      int M, int N, int K, int relu) {
  __shared__ unsigned short smA[2][128 * 40];   // padded stride 40 -> conflict-free b128
  __shared__ unsigned short smB[2][64 * 40];

  const int tid  = threadIdx.x;
  const int lane = tid & 31;
  const int wave = tid >> 5;
  const int wm = wave & 3;     // 32-row strip
  const int wn = wave >> 2;    // 32-col strip
  const int m0 = blockIdx.x * 128;
  const int n0 = blockIdx.y * 64;

  const int r  = lane & 15;
  const int hi = lane >> 4;

  // cooperative copy addressing: A = 128x32 bf16 (2 chunks/thread), B = 64x32 (1 chunk)
  const int r0 = tid >> 2;                 // 0..63
  const int c0 = (tid & 3) * 8;            // 0,8,16,24 bf16 elems
  int ag0 = m0 + r0;      if (ag0 >= M) ag0 = M - 1;
  int ag1 = m0 + r0 + 64; if (ag1 >= M) ag1 = M - 1;
  const unsigned short* aptr0 = A + (size_t)ag0 * K + c0;
  const unsigned short* aptr1 = A + (size_t)ag1 * K + c0;
  const unsigned short* bptr  = Wt + (size_t)(n0 + r0) * K + c0;

  v8f acc[2][2];
#pragma unroll
  for (int i = 0; i < 2; i++)
#pragma unroll
    for (int j = 0; j < 2; j++)
#pragma unroll
      for (int v = 0; v < 8; v++) acc[i][j][v] = 0.f;

  // prologue: stage tile 0 into buffer 0
  cp16(&smA[0][r0 * 40 + c0], aptr0);
  cp16(&smA[0][(r0 + 64) * 40 + c0], aptr1);
  cp16(&smB[0][r0 * 40 + c0], bptr);

  const int nk = K >> 5;
  for (int t = 0; t < nk; ++t) {
    const int cur = t & 1, nxt = cur ^ 1;
    const int kn = (t + 1) << 5;
    if (t + 1 < nk) {
      cp16(&smA[nxt][r0 * 40 + c0], aptr0 + kn);
      cp16(&smA[nxt][(r0 + 64) * 40 + c0], aptr1 + kn);
      cp16(&smB[nxt][r0 * 40 + c0], bptr + kn);
      __builtin_prefetch(aptr0 + kn, 0, 1);     // global_prefetch_b8
      AWAIT(3);                                 // current tile's copies done
    } else {
      AWAIT(0);
    }
    __syncthreads();

    const unsigned short* sa = smA[cur];
    const unsigned short* sb = smB[cur];

    // A fragments (16x32 bf16): lane r -> K {hi*8..+7, 16+hi*8..+7}
    v16bf av[2];
#pragma unroll
    for (int ms = 0; ms < 2; ms++) {
      v8bf a_lo = *(const v8bf*)&sa[(wm * 32 + ms * 16 + r) * 40 + hi * 8];
      v8bf a_hi = *(const v8bf*)&sa[(wm * 32 + ms * 16 + r) * 40 + 16 + hi * 8];
#pragma unroll
      for (int i = 0; i < 8; i++) { av[ms][i] = a_lo[i]; av[ms][i + 8] = a_hi[i]; }
    }
#pragma unroll
    for (int ns = 0; ns < 2; ns++) {
      // B fragment (32x16): lane group hi -> K {hi*8..+7, 16+hi*8..+7} of column r
      const int col = wn * 32 + ns * 16 + r;
      v8bf b_lo = *(const v8bf*)&sb[col * 40 + hi * 8];
      v8bf b_hi = *(const v8bf*)&sb[col * 40 + 16 + hi * 8];
      v16bf bv;
#pragma unroll
      for (int i = 0; i < 8; i++) { bv[i] = b_lo[i]; bv[i + 8] = b_hi[i]; }
#pragma unroll
      for (int ms = 0; ms < 2; ms++)
        acc[ms][ns] = __builtin_amdgcn_wmma_f32_16x16x32_bf16(false, av[ms], false, bv,
                                                              (short)0, acc[ms][ns],
                                                              false, false);
    }
    __syncthreads();
  }

  // epilogue: bias + optional relu, f32 and/or bf16 stores (2x2 tiles)
#pragma unroll
  for (int ns = 0; ns < 2; ns++) {
    const int n = n0 + wn * 32 + ns * 16 + r;
    const float bb = bias ? bias[n] : 0.f;
#pragma unroll
    for (int ms = 0; ms < 2; ms++) {
#pragma unroll
      for (int v = 0; v < 8; v++) {
        int m = m0 + wm * 32 + ms * 16 + hi * 8 + v;
        float x = acc[ms][ns][v] + bb;
        if (relu) x = fmaxf(x, 0.f);
        if (m < M) {
          if (Cf) Cf[(size_t)m * N + n] = x;
          if (Cb) Cb[(size_t)m * N + n] = f2bf(x);
        }
      }
    }
  }
}

// ---------------- deformable sampling: one wave per (query, head) -----------
__global__ __launch_bounds__(256) void msda_sample_kernel(const float* __restrict__ Vp,
                                                          const float* __restrict__ off,
                                                          const float* __restrict__ awl,
                                                          const float* __restrict__ ref,
                                                          float* __restrict__ samp,
                                                          unsigned short* __restrict__ sampb) {
  const int q    = blockIdx.x;
  const int h    = threadIdx.x >> 5;
  const int lane = threadIdx.x & 31;

  // 16-way softmax over (level,point) logits via wave shuffles
  float wlog = awl[((size_t)q * 8 + h) * 16 + (lane & 15)];
  float mx = wlog;
#pragma unroll
  for (int t = 8; t >= 1; t >>= 1) mx = fmaxf(mx, __shfl_xor(mx, t, 32));
  float e = __expf(wlog - mx);
  float sm = e;
#pragma unroll
  for (int t = 8; t >= 1; t >>= 1) sm += __shfl_xor(sm, t, 32);
  const float prob = e / sm;

  const int Hs[4]     = {92, 46, 23, 12};
  const int Ws_[4]    = {160, 80, 40, 20};
  const int starts[4] = {0, 14720, 18400, 19320};

  const float* offq = off + ((size_t)q * 8 + h) * 32;
  float acc = 0.f;

#pragma unroll
  for (int l = 0; l < 4; l++) {
    const float rx = ref[((size_t)q * 4 + l) * 2 + 0];
    const float ry = ref[((size_t)q * 4 + l) * 2 + 1];
    const int   Wl = Ws_[l], Hl = Hs[l];
    const float Wf = (float)Wl, Hf = (float)Hl;
    const float* vbase = Vp + (size_t)starts[l] * DMOD + h * 32 + lane;
#pragma unroll
    for (int p = 0; p < 4; p++) {
      const float ox = offq[(l * 4 + p) * 2 + 0];
      const float oy = offq[(l * 4 + p) * 2 + 1];
      const float x = rx * Wf + ox - 0.5f;
      const float y = ry * Hf + oy - 0.5f;
      const float x0f = floorf(x), y0f = floorf(y);
      const float lx = x - x0f, ly = y - y0f;
      const int x0 = (int)x0f, y0 = (int)y0f;
      const float aw = __shfl(prob, l * 4 + p, 32);

      float gsum = 0.f;
#pragma unroll
      for (int c = 0; c < 4; c++) {
        const int xi = x0 + (c & 1);
        const int yi = y0 + (c >> 1);
        const float wx = (c & 1) ? lx : (1.f - lx);
        const float wy = (c >> 1) ? ly : (1.f - ly);
        const bool valid = (xi >= 0) & (xi < Wl) & (yi >= 0) & (yi < Hl);
        int xc = xi < 0 ? 0 : (xi > Wl - 1 ? Wl - 1 : xi);
        int yc = yi < 0 ? 0 : (yi > Hl - 1 ? Hl - 1 : yi);
        const float v = vbase[(size_t)(yc * Wl + xc) * DMOD];   // 128B/wave coalesced
        gsum += valid ? (wx * wy) * v : 0.f;
      }
      acc += aw * gsum;
    }
  }
  const size_t o = (size_t)q * DMOD + h * 32 + lane;
  if (samp)  samp[o]  = acc;
  if (sampb) sampb[o] = f2bf(acc);
}

// ============================= host ========================================
extern "C" void kernel_launch(void* const* d_in, const int* in_sizes, int n_in,
                              void* d_out, int out_size, void* d_ws, size_t ws_size,
                              hipStream_t stream) {
  auto F = [&](int i) { return (const float*)d_in[i]; };
  const float* q    = F(0);
  const float* kvt1 = F(1);
  const float* kvt2 = F(2);
  const float* pos  = F(3);
  const float* ref  = F(4);

  const float *W_off_sa=F(7),  *b_off_sa=F(8),  *W_aw_sa=F(9),  *b_aw_sa=F(10),
              *W_v_sa=F(11),   *b_v_sa=F(12),   *W_o_sa=F(13),  *b_o_sa=F(14);
  const float *W_off_t1=F(15), *b_off_t1=F(16), *W_aw_t1=F(17), *b_aw_t1=F(18),
              *W_v_t1=F(19),   *b_v_t1=F(20),   *W_o_t1=F(21),  *b_o_t1=F(22);
  const float *W_off_t2=F(23), *b_off_t2=F(24), *W_aw_t2=F(25), *b_aw_t2=F(26),
              *W_v_t2=F(27),   *b_v_t2=F(28),   *W_o_t2=F(29),  *b_o_t2=F(30);
  const float *n0g=F(31), *n0b=F(32), *n1g=F(33), *n1b=F(34),
              *n2g=F(35), *n2b=F(36), *n5g=F(37), *n5b=F(38);
  const float *W1=F(39), *b1=F(40), *W2=F(41), *b2=F(42);

  const size_t NE = (size_t)LQc * DMOD;
  char* wp = (char*)d_ws;
  auto af = [&](size_t n) { float* p = (float*)wp; wp += n * sizeof(float); return p; };
  auto ah = [&](size_t n) { unsigned short* p = (unsigned short*)wp; wp += n * 2; return p; };

  float *Vp  = af(NE), *OFF = af(NE), *AW = af((size_t)LQc * 128),
        *TMP = af(NE), *Q1  = af(NE), *SRC2 = af(NE), *X = af(NE);
  unsigned short *QB = ah(NE), *Q1B = ah(NE), *VINB = ah(NE),
                 *SAMPB = ah(NE), *XB = ah(NE), *HB = ah((size_t)LQc * DFFN);
  unsigned short *WT_off_sa = ah(256 * 256), *WT_aw_sa = ah(256 * 128),
                 *WT_v_sa = ah(256 * 256), *WT_o_sa = ah(256 * 256);
  unsigned short *WT_off_t1 = ah(256 * 256), *WT_aw_t1 = ah(256 * 128),
                 *WT_v_t1 = ah(256 * 256), *WT_o_t1 = ah(256 * 256);
  unsigned short *WT_off_t2 = ah(256 * 256), *WT_aw_t2 = ah(256 * 128),
                 *WT_v_t2 = ah(256 * 256), *WT_o_t2 = ah(256 * 256);
  unsigned short *WT1 = ah(256 * 1024), *WT2 = ah(1024 * 256);

  float* OUT = (float*)d_out;          // [src3 | q_out]

  auto wcvt = [&](const float* W, unsigned short* Wt, int K, int N) {
    int n = K * N;
    wt_cvt_kernel<<<(n + 255) / 256, 256, 0, stream>>>(W, Wt, K, N);
  };
  auto gemm = [&](const unsigned short* A, const unsigned short* Wt, const float* bias,
                  float* Cf, unsigned short* Cb, int M, int N, int K, int relu) {
    dim3 g((M + 127) / 128, N / 64);
    gemm_bf16_wmma<<<g, 256, 0, stream>>>(A, Wt, bias, Cf, Cb, M, N, K, relu);
  };
  auto addc = [&](const float* a, const float* b, float* of, unsigned short* oh, size_t n) {
    int n4 = (int)(n / 4);
    add_cvt_kernel<<<(n4 + 255) / 256, 256, 0, stream>>>(a, b, of, oh, n4);
  };
  auto ln = [&](const float* a, const float* resid, const float* g, const float* b,
                float* out, unsigned short* outb) {
    ln_kernel<<<(LQc + 7) / 8, 256, 0, stream>>>(a, resid, g, b, out, outb, LQc);
  };
  auto msda = [&](const float* vp, const float* off_, const float* aw_) {
    msda_sample_kernel<<<LQc, 256, 0, stream>>>(vp, off_, aw_, ref, nullptr, SAMPB);
  };

  // ---- weight repack to bf16 [N,K] ----
  wcvt(W_off_sa, WT_off_sa, 256, 256); wcvt(W_aw_sa, WT_aw_sa, 256, 128);
  wcvt(W_v_sa,   WT_v_sa,   256, 256); wcvt(W_o_sa,  WT_o_sa,  256, 256);
  wcvt(W_off_t1, WT_off_t1, 256, 256); wcvt(W_aw_t1, WT_aw_t1, 256, 128);
  wcvt(W_v_t1,   WT_v_t1,   256, 256); wcvt(W_o_t1,  WT_o_t1,  256, 256);
  wcvt(W_off_t2, WT_off_t2, 256, 256); wcvt(W_aw_t2, WT_aw_t2, 256, 128);
  wcvt(W_v_t2,   WT_v_t2,   256, 256); wcvt(W_o_t2,  WT_o_t2,  256, 256);
  wcvt(W1, WT1, 256, 1024);            wcvt(W2, WT2, 1024, 256);

  addc(q, nullptr, nullptr, QB, NE);   // q -> bf16

  // ---- self-attention MSDA ----
  gemm(QB, WT_v_sa,   b_v_sa,   Vp,  nullptr, LQc, 256, 256, 0);
  gemm(QB, WT_off_sa, b_off_sa, OFF, nullptr, LQc, 256, 256, 0);
  gemm(QB, WT_aw_sa,  b_aw_sa,  AW,  nullptr, LQc, 128, 256, 0);
  msda(Vp, OFF, AW);
  gemm(SAMPB, WT_o_sa, b_o_sa, TMP, nullptr, LQc, 256, 256, 0);
  ln(q, TMP, n0g, n0b, Q1, Q1B);                       // q1 (+bf16)

  // ---- cross-attention T1 ----
  addc(kvt1, pos, nullptr, VINB, NE);
  gemm(VINB, WT_v_t1,   b_v_t1,   Vp,  nullptr, LQc, 256, 256, 0);
  gemm(Q1B,  WT_off_t1, b_off_t1, OFF, nullptr, LQc, 256, 256, 0);
  gemm(Q1B,  WT_aw_t1,  b_aw_t1,  AW,  nullptr, LQc, 128, 256, 0);
  msda(Vp, OFF, AW);
  gemm(SAMPB, WT_o_t1, b_o_t1, TMP, nullptr, LQc, 256, 256, 0);
  ln(TMP, nullptr, n1g, n1b, SRC2, nullptr);           // src2

  // ---- cross-attention T2 ----
  addc(kvt2, pos, nullptr, VINB, NE);
  gemm(VINB, WT_v_t2,   b_v_t2,   Vp,  nullptr, LQc, 256, 256, 0);
  gemm(Q1B,  WT_off_t2, b_off_t2, OFF, nullptr, LQc, 256, 256, 0);
  gemm(Q1B,  WT_aw_t2,  b_aw_t2,  AW,  nullptr, LQc, 128, 256, 0);
  msda(Vp, OFF, AW);
  gemm(SAMPB, WT_o_t2, b_o_t2, TMP, nullptr, LQc, 256, 256, 0);
  ln(SRC2, TMP, n2g, n2b, OUT, nullptr);               // src3 -> output[0]

  // ---- FFN ----
  addc(Q1, OUT, X, XB, NE);                            // x = q1 + src3 (+bf16)
  gemm(XB, WT1, b1, nullptr, HB, LQc, 1024, 256, 1);   // h = relu(x@W1+b1) -> bf16
  gemm(HB, WT2, b2, TMP, nullptr, LQc, 256, 1024, 0);  // ffn = h@W2+b2
  ln(X, TMP, n5g, n5b, OUT + NE, nullptr);             // q_out -> output[1]
}
